// MatchingLayer_1778116461332
// MI455X (gfx1250) — matrix operand order repaired and checked by reading, not compiled, not run on hardware
//
#include <hip/hip_runtime.h>
#include <math.h>

#define B_  8
#define L_  128
#define H_  768
#define P_  256
#define BP  2048          // B_ * P_
#define FH  (5 * H_)      // 3840 features per pair
#define HQ  (H_ / 4)      // 192 float4 per H-row

typedef _Float16 h4   __attribute__((ext_vector_type(4)));
typedef _Float16 h8   __attribute__((ext_vector_type(8)));
typedef _Float16 v16h __attribute__((ext_vector_type(16)));
typedef float    v8f  __attribute__((ext_vector_type(8)));

__device__ inline void st4h(_Float16* p, float4 v) {
    h4 x = { (_Float16)v.x, (_Float16)v.y, (_Float16)v.z, (_Float16)v.w };
    *(h4*)p = x;
}

// ---------------------------------------------------------------------------
// Kernel 1: per-pair feature encoding (gathers + dynamic-window max-pool).
// 1 block per pair, 192 threads, each owning one float4 of the H dimension.
// Fully coalesced 3KB row-chunks; bandwidth-bound (the roofline limiter).
// ---------------------------------------------------------------------------
__global__ __launch_bounds__(192) void feats_kernel(
    const float* __restrict__ Table, const float* __restrict__ Edge,
    const int* __restrict__ pairs, _Float16* __restrict__ feats)
{
    const int p  = blockIdx.x;      // pair index 0..2047
    const int bi = p / P_;
    const int t  = threadIdx.x;     // 0..191 -> h = 4t..4t+3

    const int4 pr = ((const int4*)pairs)[p];
    const int s0 = pr.x, e0 = pr.y, s1 = pr.z, e1 = pr.w;
    const bool asp = (s0 < 0);
    const int r0 = asp ? s1 : s0;
    const int q0 = asp ? e1 : e0;

    const size_t base4 = (size_t)bi * ((size_t)L_ * L_ * HQ);
    const float4* Tb = (const float4*)Table + base4;
    const float4* Eb = (const float4*)Edge  + base4;

    // Single-row gathers
    float4 S  = Tb[(size_t)((r0 + 1) * L_ + (s1 + 1)) * HQ + t];
    float4 Ev = Tb[(size_t)(q0 * L_ + e1) * HQ + t];
    float4 S2 = Eb[(size_t)((s0 + 1) * L_ + (s1 + 1)) * HQ + t];
    const int rowE2 = (e0 < 0) ? (L_ - 1) : e0;   // python % semantics for -1
    float4 E2 = Eb[(size_t)(rowE2 * L_ + e1) * HQ + t];

    // Dynamic-window max-pool over rows [r0+1, q0], cols [s1+1, e1]
    float4 R = make_float4(-INFINITY, -INFINITY, -INFINITY, -INFINITY);
    for (int i = r0 + 1; i <= q0; ++i) {
        const float4* rowp = Tb + (size_t)(i * L_) * HQ + t;
        for (int j = s1 + 1; j <= e1; ++j) {
            float4 v = rowp[(size_t)j * HQ];
            R.x = fmaxf(R.x, v.x); R.y = fmaxf(R.y, v.y);
            R.z = fmaxf(R.z, v.z); R.w = fmaxf(R.w, v.w);
        }
    }

    _Float16* f = feats + (size_t)p * FH + 4 * t;
    st4h(f + 0 * H_, S);
    st4h(f + 1 * H_, Ev);
    st4h(f + 2 * H_, R);
    st4h(f + 3 * H_, S2);
    st4h(f + 4 * H_, E2);
}

// ---------------------------------------------------------------------------
// Kernel 2: logits via v_wmma_f32_16x16x32_f16 (16 pairs per wave, K=3840 ->
// 120 WMMA per wave), then softmax / probs / NLL.
// 32 blocks x 128 threads (4 waves). W^T is zero-padded to a full 16-column
// fp16 B-matrix in LDS once, so the inner loop is pure load+WMMA (no masking).
// ---------------------------------------------------------------------------
__global__ __launch_bounds__(128) void logits_kernel(
    const _Float16* __restrict__ feats, const float* __restrict__ Wmat,
    const float* __restrict__ bias, const int* __restrict__ labels,
    float* __restrict__ probs, float* __restrict__ nll,
    float* __restrict__ validf)
{
    __shared__ _Float16 Wh[16 * FH];      // 120 KB: W as fp16, rows 4..15 zero
    __shared__ float    lg[4][16][4];     // per-wave logits exchange

    const int tid = threadIdx.x;
    for (int i = tid; i < 16 * FH; i += 128)
        Wh[i] = (i < 4 * FH) ? (_Float16)Wmat[i] : (_Float16)0.f;
    __syncthreads();

    const int w    = tid >> 5;            // wave 0..3
    const int lane = tid & 31;
    const int n    = lane & 15;           // A: row M / B,C,D: col N
    const int g    = lane >> 4;           // lane half-group

    const int pg       = blockIdx.x * 4 + w;   // pair group 0..127
    const int pairRow  = pg * 16 + n;          // A-matrix row M = pair
    const _Float16* fa = feats + (size_t)pairRow * FH;
    const _Float16* wb = Wh + (size_t)n * FH + 8 * g;

    v8f acc = {0.f, 0.f, 0.f, 0.f, 0.f, 0.f, 0.f, 0.f};

    #pragma unroll 4
    for (int kb = 0; kb < FH; kb += 32) {
        // A fragment: 16-bit A 16x32 layout. Lane group g: K = 8g..8g+7 then 16+8g..23+8g.
        h8 alo = *(const h8*)(fa + kb + 8 * g);
        h8 ahi = *(const h8*)(fa + kb + 16 + 8 * g);
        v16h a;
        #pragma unroll
        for (int i = 0; i < 8; ++i) { a[i] = alo[i]; a[8 + i] = ahi[i]; }

        // B fragment: 32x16, lane group g: K = 8g..8g+7 then 16+8g..23+8g. B[k][n] = Wh[n][kb+k].
        h8 blo = *(const h8*)(wb + kb);
        h8 bhi = *(const h8*)(wb + kb + 16);
        v16h bm;
        #pragma unroll
        for (int i = 0; i < 8; ++i) { bm[i] = blo[i]; bm[8 + i] = bhi[i]; }

        acc = __builtin_amdgcn_wmma_f32_16x16x32_f16(
            false, a, false, bm, (short)0, acc, false, false);
    }

    // D layout: VGPR r, lane-group g -> M = r + 8g, N = n. Stash logits in LDS.
    if (n < 4) {
        float bc = bias[n];
        #pragma unroll
        for (int r = 0; r < 8; ++r) lg[w][r + 8 * g][n] = acc[r] + bc;
    }
    __syncthreads();

    if (lane < 16) {
        const int pp = pg * 16 + lane;
        float l0 = lg[w][lane][0], l1 = lg[w][lane][1];
        float l2 = lg[w][lane][2], l3 = lg[w][lane][3];
        float m  = fmaxf(fmaxf(l0, l1), fmaxf(l2, l3));
        float x0 = __expf(l0 - m), x1 = __expf(l1 - m);
        float x2 = __expf(l2 - m), x3 = __expf(l3 - m);
        float s   = x0 + x1 + x2 + x3;
        float inv = 1.f / s;
        probs[pp * 4 + 0] = x0 * inv;
        probs[pp * 4 + 1] = x1 * inv;
        probs[pp * 4 + 2] = x2 * inv;
        probs[pp * 4 + 3] = x3 * inv;

        int  lab   = labels[pp];
        bool valid = (lab >= 0);
        int  lc    = valid ? lab : 0;
        float ll   = (lc == 0) ? l0 : (lc == 1) ? l1 : (lc == 2) ? l2 : l3;
        float logp = (ll - m) - __logf(s);
        nll[pp]    = valid ? -logp : 0.f;
        validf[pp] = valid ? 1.f : 0.f;
    }
}

// ---------------------------------------------------------------------------
// Kernel 3: deterministic single-block reduction -> loss.
// ---------------------------------------------------------------------------
__global__ __launch_bounds__(256) void loss_kernel(
    const float* __restrict__ nll, const float* __restrict__ validf,
    float* __restrict__ out)
{
    __shared__ float sn[256], sv[256];
    const int t = threadIdx.x;
    float a = 0.f, c = 0.f;
    for (int i = t; i < BP; i += 256) { a += nll[i]; c += validf[i]; }
    sn[t] = a; sv[t] = c;
    __syncthreads();
    for (int s = 128; s > 0; s >>= 1) {
        if (t < s) { sn[t] += sn[t + s]; sv[t] += sv[t + s]; }
        __syncthreads();
    }
    if (t == 0) out[0] = sn[0] / fmaxf(sv[0], 1.f);
}

// ---------------------------------------------------------------------------
extern "C" void kernel_launch(void* const* d_in, const int* in_sizes, int n_in,
                              void* d_out, int out_size, void* d_ws, size_t ws_size,
                              hipStream_t stream)
{
    (void)in_sizes; (void)n_in; (void)out_size; (void)ws_size;

    const float* Table = (const float*)d_in[0];
    const float* Edge  = (const float*)d_in[1];
    const int*   pairs = (const int*)d_in[2];
    const int*   labels= (const int*)d_in[3];
    const float* Wmat  = (const float*)d_in[4];
    const float* bias  = (const float*)d_in[5];

    float* out = (float*)d_out;               // [0]=loss, [1..8192]=probs

    _Float16* feats  = (_Float16*)d_ws;                                   // 15.7 MB
    float*    nll    = (float*)((char*)d_ws + (size_t)BP * FH * sizeof(_Float16));
    float*    validf = nll + BP;

    feats_kernel <<<BP, 192, 0, stream>>>(Table, Edge, pairs, feats);
    logits_kernel<<<BP / 64, 128, 0, stream>>>(feats, Wmat, bias, labels,
                                               out + 1, nll, validf);
    loss_kernel  <<<1, 256, 0, stream>>>(nll, validf, out);
}
